// ImpactFunction_87351044866338
// MI455X (gfx1250) — matrix operand order repaired and checked by reading, not compiled, not run on hardware
//
#include <hip/hip_runtime.h>
#include <hip/hip_bf16.h>
#include <math.h>

#define N_NODES 50000
#define N_EDGES 800000
#define HID 256
#define TDIM 128
#define MT (N_NODES / 16)   // 3125 row tiles

typedef __attribute__((ext_vector_type(16))) __bf16 v16bf;
typedef __attribute__((ext_vector_type(8)))  float  v8f;

__device__ __forceinline__ unsigned short f2bf(float f) {
  unsigned u = __float_as_uint(f);
  u += 0x7fffu + ((u >> 16) & 1u);          // round-to-nearest-even
  return (unsigned short)(u >> 16);
}

__device__ __forceinline__ v8f wmma_bf16(v16bf a, v16bf b, v8f c) {
  // D = A(16x32 bf16) * B(32x16 bf16) + C(16x16 f32)
  return __builtin_amdgcn_wmma_f32_16x16x32_bf16(false, a, false, b, (short)0, c,
                                                 false, false);
}

// ---------------------------------------------------------------- utilities
__global__ void k_zero(unsigned int* p, int n) {
  int i = blockIdx.x * blockDim.x + threadIdx.x;
  if (i < n) p[i] = 0u;
}

__global__ void k_degree(const int* __restrict__ src, const int* __restrict__ dst,
                         unsigned int* dout, unsigned int* din) {
  int e = blockIdx.x * blockDim.x + threadIdx.x;
  if (e < N_EDGES) {
    atomicAdd(&dout[src[e]], 1u);
    atomicAdd(&din[dst[e]], 1u);
  }
}

__global__ void k_norm(unsigned int* deg, float* norm) {  // in-place reinterpret
  int i = blockIdx.x * blockDim.x + threadIdx.x;
  if (i < N_NODES) {
    unsigned d = deg[i];
    float f = (float)(d < 1u ? 1u : d);
    norm[i] = rsqrtf(f);
  }
}

// one wave32 per edge: gather h[src]*out_norm, atomic-add into agg[dst]
__global__ __launch_bounds__(256) void k_scatter(const float* __restrict__ x,
                                                 const float* __restrict__ outn,
                                                 const int* __restrict__ src,
                                                 const int* __restrict__ dst,
                                                 float* agg) {
  int e = blockIdx.x * 8 + (threadIdx.x >> 5);
  int lane = threadIdx.x & 31;
  int s = src[e], d = dst[e];
  float on = outn[s];
  const float* xs = x + (size_t)s * HID;
  float* ad = agg + (size_t)d * HID;
#pragma unroll
  for (int i = 0; i < 8; ++i) {
    int k = lane + i * 32;
    unsafeAtomicAdd(&ad[k], xs[k] * on);    // global_atomic_add_f32
  }
}

// ------------------------------------------------- fragment-layout packing
// A fragment (16x32 bf16): lane m=lane&15, kbase=(lane>=16)?8:0,
//   frag[j] = A[m][kbase + (j<8 ? j : j+8)]   (16 bf16 contiguous per lane)
__global__ void k_pack_a(const float* __restrict__ x, const float* __restrict__ innorm,
                         unsigned short* dst, int KT) {
  int t = blockIdx.x * blockDim.x + threadIdx.x;
  if (t >= MT * KT * 512) return;
  int j = t & 15, lane = (t >> 4) & 31, rest = t >> 9;
  int kt = rest % KT, mt = rest / KT;
  int m = mt * 16 + (lane & 15);
  int k = kt * 32 + ((lane & 16) >> 1) + (j < 8 ? j : j + 8);
  dst[t] = f2bf(x[(size_t)m * HID + k] * innorm[m]);
}

// x = [h | cos(t*w+b)]  packed for K=384 (12 k-tiles)
__global__ void k_pack_x(const float* __restrict__ h, const float* __restrict__ tptr,
                         const float* __restrict__ tw, const float* __restrict__ tb,
                         unsigned short* dst) {
  int t = blockIdx.x * blockDim.x + threadIdx.x;
  if (t >= MT * 12 * 512) return;
  int j = t & 15, lane = (t >> 4) & 31, rest = t >> 9;
  int kt = rest % 12, mt = rest / 12;
  int m = mt * 16 + (lane & 15);
  int k = kt * 32 + ((lane & 16) >> 1) + (j < 8 ? j : j + 8);
  float v;
  if (k < HID) v = h[(size_t)m * HID + k];
  else { int kk = k - HID; v = cosf(tptr[0] * tw[kk] + tb[kk]); }
  dst[t] = f2bf(v);
}

// B fragment (32x16 bf16): n=lane&15, kbase=(lane>=16)?8:0, frag[j]=B[kbase + (j<8 ? j : j+8)][n]
// transpose=0: B[k][n]=W[k*Nout+n] (W is [in,out]); transpose=1: B[k][n]=W[n*K+k]
__global__ void k_pack_b(const float* __restrict__ W, unsigned short* dst,
                         int K, int Nout, int transpose) {
  int t = blockIdx.x * blockDim.x + threadIdx.x;
  if (t >= K * Nout) return;
  int NT = Nout >> 4;
  int j = t & 15, lane = (t >> 4) & 31, rest = t >> 9;
  int nt = rest % NT, kt = rest / NT;
  int k = kt * 32 + ((lane & 16) >> 1) + (j < 8 ? j : j + 8);
  int n = nt * 16 + (lane & 15);
  float v = transpose ? W[(size_t)n * K + k] : W[(size_t)k * Nout + n];
  dst[t] = f2bf(v);
}

// ---------------------------------------------------------------- GCN GEMM
// out[M,256] = relu(A[M,256] @ B[256,256] + bias); 4 waves/block, 16x64 per wave
__global__ __launch_bounds__(128) void k_gcn_gemm(const unsigned short* __restrict__ Ap,
                                                  const unsigned short* __restrict__ Bp,
                                                  const float* __restrict__ bias,
                                                  float* __restrict__ out) {
  const int mt = blockIdx.x;
  const int lane = threadIdx.x & 31;
  const int wv = threadIdx.x >> 5;
  const int c0 = wv * 64;
  v8f acc[4];
#pragma unroll
  for (int i = 0; i < 4; ++i) acc[i] = (v8f){0.f, 0.f, 0.f, 0.f, 0.f, 0.f, 0.f, 0.f};
#pragma unroll
  for (int kt = 0; kt < 8; ++kt) {
    const unsigned short* ap = Ap + ((size_t)(mt * 8 + kt) * 32 + lane) * 16;
    if (kt < 7) __builtin_prefetch(ap + 512 * 16, 0, 3);
    v16bf a = *(const v16bf*)ap;
#pragma unroll
    for (int nt = 0; nt < 4; ++nt) {
      int ntG = (c0 >> 4) + nt;
      v16bf b = *(const v16bf*)(Bp + ((size_t)(kt * 16 + ntG) * 32 + lane) * 16);
      acc[nt] = wmma_bf16(a, b, acc[nt]);
    }
  }
  const int mrow = mt * 16 + ((lane & 16) >> 1);  // +8 for upper half-wave
  const int ncol = lane & 15;
#pragma unroll
  for (int nt = 0; nt < 4; ++nt) {
    int col = c0 + nt * 16 + ncol;
    float bv = bias[col];
#pragma unroll
    for (int r = 0; r < 8; ++r) {
      float v = fmaxf(acc[nt][r] + bv, 0.f);
      out[(size_t)(mrow + r) * HID + col] = v;
    }
  }
}

// ------------------------------------------------------- fused GRU (no gi/gh)
// wave: 16 rows x 32 gate-cols; 6 gate accumulators (gi_r,z,n / gh_r,z,n) x 2 tiles
__global__ __launch_bounds__(256) void k_gru(const unsigned short* __restrict__ Ap,
                                             const unsigned short* __restrict__ Bih,
                                             const unsigned short* __restrict__ Bhh,
                                             const float* __restrict__ b_ih,
                                             const float* __restrict__ b_hh,
                                             const float* __restrict__ h,
                                             float* __restrict__ out) {
  const int mt = blockIdx.x;
  const int lane = threadIdx.x & 31;
  const int wv = threadIdx.x >> 5;   // 0..7
  const int c0 = wv * 32;
  v8f acc[6][2];
#pragma unroll
  for (int g = 0; g < 6; ++g)
#pragma unroll
    for (int nt = 0; nt < 2; ++nt)
      acc[g][nt] = (v8f){0.f, 0.f, 0.f, 0.f, 0.f, 0.f, 0.f, 0.f};

#pragma unroll
  for (int kt = 0; kt < 12; ++kt) {
    v16bf a = *(const v16bf*)(Ap + ((size_t)(mt * 12 + kt) * 32 + lane) * 16);
#pragma unroll
    for (int g = 0; g < 3; ++g)
#pragma unroll
      for (int nt = 0; nt < 2; ++nt) {
        int ntG = g * 16 + (c0 >> 4) + nt;           // Nout=768 -> NT=48
        v16bf b = *(const v16bf*)(Bih + ((size_t)(kt * 48 + ntG) * 32 + lane) * 16);
        acc[g][nt] = wmma_bf16(a, b, acc[g][nt]);
      }
    if (kt < 8) {                                    // gh: K=256 only
#pragma unroll
      for (int g = 0; g < 3; ++g)
#pragma unroll
        for (int nt = 0; nt < 2; ++nt) {
          int ntG = g * 16 + (c0 >> 4) + nt;
          v16bf b = *(const v16bf*)(Bhh + ((size_t)(kt * 48 + ntG) * 32 + lane) * 16);
          acc[3 + g][nt] = wmma_bf16(a, b, acc[3 + g][nt]);
        }
    }
  }
  const int mrow0 = mt * 16 + ((lane & 16) >> 1);
  const int ncol = lane & 15;
#pragma unroll
  for (int nt = 0; nt < 2; ++nt) {
    int j = c0 + nt * 16 + ncol;
    float bir = b_ih[j], biz = b_ih[256 + j], bin = b_ih[512 + j];
    float bhr = b_hh[j], bhz = b_hh[256 + j], bhn = b_hh[512 + j];
#pragma unroll
    for (int r = 0; r < 8; ++r) {
      int m = mrow0 + r;
      float gir = acc[0][nt][r] + bir, giz = acc[1][nt][r] + biz, gin = acc[2][nt][r] + bin;
      float ghr = acc[3][nt][r] + bhr, ghz = acc[4][nt][r] + bhz, ghn = acc[5][nt][r] + bhn;
      float rg = 1.f / (1.f + expf(-(gir + ghr)));
      float z  = 1.f / (1.f + expf(-(giz + ghz)));
      float nn = tanhf(gin + rg * ghn);
      float hv = h[(size_t)m * HID + j];
      out[(size_t)m * HID + j] = (1.f - z) * nn + z * hv;
    }
  }
}

// ---------------------------------------------------------------- launcher
extern "C" void kernel_launch(void* const* d_in, const int* in_sizes, int n_in,
                              void* d_out, int out_size, void* d_ws, size_t ws_size,
                              hipStream_t stream) {
  const float* hidden = (const float*)d_in[0];
  const float* t      = (const float*)d_in[1];
  const int*   src    = (const int*)d_in[2];
  const int*   dst    = (const int*)d_in[3];
  const float* W1   = (const float*)d_in[4];
  const float* b1   = (const float*)d_in[5];
  const float* W2   = (const float*)d_in[6];
  const float* b2   = (const float*)d_in[7];
  const float* tw   = (const float*)d_in[8];
  const float* tb   = (const float*)d_in[9];
  const float* W_ih = (const float*)d_in[10];
  const float* W_hh = (const float*)d_in[11];
  const float* b_ih = (const float*)d_in[12];
  const float* b_hh = (const float*)d_in[13];
  float* out = (float*)d_out;
  (void)in_sizes; (void)n_in; (void)out_size; (void)ws_size;

  char* ws = (char*)d_ws;
  size_t off = 0;
  auto take = [&](size_t bytes) {
    char* p = ws + off;
    off += (bytes + 255) & ~(size_t)255;
    return p;
  };
  unsigned* degOut = (unsigned*)take((size_t)N_NODES * 4);
  unsigned* degIn  = (unsigned*)take((size_t)N_NODES * 4);
  float* normOut = (float*)degOut;   // converted in place
  float* normIn  = (float*)degIn;
  float* bufA = (float*)take((size_t)N_NODES * HID * 4);   // agg scratch
  float* bufB = (float*)take((size_t)N_NODES * HID * 4);   // h
  unsigned short* Ap   = (unsigned short*)take((size_t)MT * 12 * 512 * 2); // packed A
  unsigned short* W1p  = (unsigned short*)take((size_t)HID * HID * 2);
  unsigned short* W2p  = (unsigned short*)take((size_t)HID * HID * 2);
  unsigned short* Bihp = (unsigned short*)take((size_t)384 * 768 * 2);
  unsigned short* Bhhp = (unsigned short*)take((size_t)256 * 768 * 2);

  const int B = 256;
  // degrees -> norms
  k_zero<<<(N_NODES + B - 1) / B, B, 0, stream>>>(degOut, N_NODES);
  k_zero<<<(N_NODES + B - 1) / B, B, 0, stream>>>(degIn, N_NODES);
  k_degree<<<(N_EDGES + B - 1) / B, B, 0, stream>>>(src, dst, degOut, degIn);
  k_norm<<<(N_NODES + B - 1) / B, B, 0, stream>>>(degOut, normOut);
  k_norm<<<(N_NODES + B - 1) / B, B, 0, stream>>>(degIn, normIn);
  // pack weights (bf16, WMMA B-fragment order)
  k_pack_b<<<(HID * HID + B - 1) / B, B, 0, stream>>>(W1, W1p, HID, HID, 0);
  k_pack_b<<<(HID * HID + B - 1) / B, B, 0, stream>>>(W2, W2p, HID, HID, 0);
  k_pack_b<<<(384 * 768 + B - 1) / B, B, 0, stream>>>(W_ih, Bihp, 384, 768, 1);
  k_pack_b<<<(256 * 768 + B - 1) / B, B, 0, stream>>>(W_hh, Bhhp, 256, 768, 1);
  // GCN layer 1
  k_zero<<<((N_NODES * HID) + B - 1) / B, B, 0, stream>>>((unsigned*)bufA, N_NODES * HID);
  k_scatter<<<N_EDGES / 8, B, 0, stream>>>(hidden, normOut, src, dst, bufA);
  k_pack_a<<<(MT * 8 * 512) / B, B, 0, stream>>>(bufA, normIn, Ap, 8);
  k_gcn_gemm<<<MT, 128, 0, stream>>>(Ap, W1p, b1, bufB);
  // GCN layer 2 (+relu, matching reference's post-relu)
  k_zero<<<((N_NODES * HID) + B - 1) / B, B, 0, stream>>>((unsigned*)bufA, N_NODES * HID);
  k_scatter<<<N_EDGES / 8, B, 0, stream>>>(bufB, normOut, src, dst, bufA);
  k_pack_a<<<(MT * 8 * 512) / B, B, 0, stream>>>(bufA, normIn, Ap, 8);
  k_gcn_gemm<<<MT, 128, 0, stream>>>(Ap, W2p, b2, bufB);
  // x = [h | time-encoding], then fused GRU (gi+gh GEMMs + gates in epilogue)
  k_pack_x<<<(MT * 12 * 512) / B, B, 0, stream>>>(bufB, t, tw, tb, Ap);
  k_gru<<<MT, B, 0, stream>>>(Ap, Bihp, Bhhp, b_ih, b_hh, bufB, out);
}
